// EuclideanDistanceAttention_69303592288272
// MI455X (gfx1250) — matrix operand order repaired and checked by reading, not compiled, not run on hardware
//
#include <hip/hip_runtime.h>
#include <hip/hip_bf16.h>
#include <math.h>

// Problem dims (fixed by the reference)
constexpr int B  = 8;
constexpr int SQ = 2048;
constexpr int SK = 2048;
constexpr int D  = 512;

typedef __attribute__((ext_vector_type(16))) _Float16 v16h;
typedef __attribute__((ext_vector_type(8)))  _Float16 v8h;
typedef __attribute__((ext_vector_type(4)))  _Float16 v4h;
typedef __attribute__((ext_vector_type(8)))  float    v8f;

__device__ __forceinline__ v8f wmma_f16(v16h a, v16h b, v8f c) {
  return __builtin_amdgcn_wmma_f32_16x16x32_f16(false, a, false, b, (short)0, c,
                                                false, false);
}

__device__ __forceinline__ v16h cat16(v8h lo, v8h hi) {
  return __builtin_shufflevector(lo, hi, 0, 1, 2, 3, 4, 5, 6, 7,
                                 8, 9, 10, 11, 12, 13, 14, 15);
}

// ---------------------------------------------------------------------------
// Prep A: split fp32 -> (hi, lo) f16 pair, elementwise. hi+lo ~ 22-bit mantissa.
// ---------------------------------------------------------------------------
__global__ __launch_bounds__(256) void prep_split(const float* __restrict__ src,
                                                  _Float16* __restrict__ hi,
                                                  _Float16* __restrict__ lo) {
  long i = (long)blockIdx.x * 256 + threadIdx.x;   // one float4 per thread
  float4 x = ((const float4*)src)[i];
  v4h h, l;
  float xs[4] = {x.x, x.y, x.z, x.w};
#pragma unroll
  for (int e = 0; e < 4; ++e) {
    _Float16 hh = (_Float16)xs[e];
    h[e] = hh;
    l[e] = (_Float16)(xs[e] - (float)hh);
  }
  *(v4h*)(hi + i * 4) = h;
  *(v4h*)(lo + i * 4) = l;
}

// ---------------------------------------------------------------------------
// Prep B: V [b][k][n] fp32 -> v_t [b][n][k] f16 (transposed for P.V B-frags)
// ---------------------------------------------------------------------------
__global__ __launch_bounds__(256) void prep_vt(const float* __restrict__ v,
                                               _Float16* __restrict__ vt) {
  long idx = (long)blockIdx.x * 256 + threadIdx.x;   // one float4 (4 n) per thread
  const int NW = D / 4;                              // 128 float4 per row
  long row = idx / NW;                               // b*SK + k
  int n4   = (int)(idx % NW);
  int b = (int)(row / SK);
  int k = (int)(row % SK);
  float4 x = ((const float4*)v)[idx];
  _Float16* dst = vt + ((long)b * D + n4 * 4) * SK + k;
  dst[0L * SK] = (_Float16)x.x;
  dst[1L * SK] = (_Float16)x.y;
  dst[2L * SK] = (_Float16)x.z;
  dst[3L * SK] = (_Float16)x.w;
}

// ---------------------------------------------------------------------------
// Prep C: ksq[j] = sum_d key[j,d]^2
// ---------------------------------------------------------------------------
__global__ __launch_bounds__(256) void k0_ksq(const float* __restrict__ key,
                                              float* __restrict__ ksq) {
  long j = (long)blockIdx.x * 256 + threadIdx.x;     // 0 .. B*SK-1
  const float4* p = (const float4*)(key + j * (long)D);
  float s = 0.f;
#pragma unroll 4
  for (int i = 0; i < D / 4; ++i) {
    float4 x = p[i];
    s += x.x * x.x + x.y * x.y + x.z * x.z + x.w * x.w;
  }
  ksq[j] = s;
}

// ---------------------------------------------------------------------------
// Kernel 1: raw logits L[b,i,j] = (2*q_i.k_j - ||k_j||^2) / T
// Split-f16 WMMA (qk ~ fp32 accurate); wave computes a 32x32 tile (2x2 frags).
// ---------------------------------------------------------------------------
__global__ __launch_bounds__(256) void k1_logits(
    const _Float16* __restrict__ qh, const _Float16* __restrict__ ql,
    const _Float16* __restrict__ kh, const _Float16* __restrict__ kl,
    const float* __restrict__ ksq, const float* __restrict__ tptr,
    float* __restrict__ logits) {
  const int TI = SQ / 32, TJ = SK / 32;            // 64 x 64 tiles / batch
  const int wave = threadIdx.x >> 5;
  const int lane = threadIdx.x & 31;
  const long tile = (long)blockIdx.x * 8 + wave;
  const int b   = (int)(tile / (TI * TJ));
  const int rem = (int)(tile % (TI * TJ));
  const int ti = rem / TJ, tj = rem % TJ;

  const int m  = lane & 15;
  const int hs = lane >> 4;

  const long qrow = ((long)b * SQ + ti * 32 + m) * D;   // + mi*16*D
  const long krow = ((long)b * SK + tj * 32 + m) * D;   // + ni*16*D
  const _Float16* qhb = qh + qrow;
  const _Float16* qlb = ql + qrow;
  const _Float16* khb = kh + krow;
  const _Float16* klb = kl + krow;

  v8f acc[2][2] = {};
  for (int kk = 0; kk < D; kk += 32) {
    v16h ah[2], al[2], bh[2], bl[2];
#pragma unroll
    for (int mi = 0; mi < 2; ++mi) {
      const _Float16* p = qhb + (long)mi * 16 * D + kk + hs * 8;
      ah[mi] = cat16(*(const v8h*)p, *(const v8h*)(p + 16));
      const _Float16* pl = qlb + (long)mi * 16 * D + kk + hs * 8;
      al[mi] = cat16(*(const v8h*)pl, *(const v8h*)(pl + 16));
    }
#pragma unroll
    for (int ni = 0; ni < 2; ++ni) {
      const _Float16* q = khb + (long)ni * 16 * D + kk + hs * 8;
      bh[ni] = cat16(*(const v8h*)q, *(const v8h*)(q + 16));
      const _Float16* ql2 = klb + (long)ni * 16 * D + kk + hs * 8;
      bl[ni] = cat16(*(const v8h*)ql2, *(const v8h*)(ql2 + 16));
    }
#pragma unroll
    for (int mi = 0; mi < 2; ++mi)
#pragma unroll
      for (int ni = 0; ni < 2; ++ni) {
        acc[mi][ni] = wmma_f16(ah[mi], bh[ni], acc[mi][ni]);
        acc[mi][ni] = wmma_f16(ah[mi], bl[ni], acc[mi][ni]);
        acc[mi][ni] = wmma_f16(al[mi], bh[ni], acc[mi][ni]);
      }
  }

  // Epilogue: C/D layout -> col n = lane&15, VGPR r -> row r + 8*hs
  const float invT = 1.0f / tptr[0];
#pragma unroll
  for (int ni = 0; ni < 2; ++ni) {
    const float ks = ksq[(long)b * SK + tj * 32 + ni * 16 + m];
#pragma unroll
    for (int mi = 0; mi < 2; ++mi) {
      float* orow = logits + ((long)b * SQ + ti * 32 + mi * 16) * SK +
                    (long)tj * 32 + ni * 16 + m;
#pragma unroll
      for (int r = 0; r < 8; ++r) {
        int mm = r + 8 * hs;
        orow[(long)mm * SK] = (2.0f * acc[mi][ni][r] - ks) * invT;
      }
    }
  }
}

// ---------------------------------------------------------------------------
// Kernel 2: in-place row softmax over Sk=2048; also emits f16 copy of weights.
// ---------------------------------------------------------------------------
__global__ __launch_bounds__(256) void k2_softmax(float* __restrict__ wts,
                                                  _Float16* __restrict__ w16) {
  __shared__ float red[8];
  const long row = blockIdx.x;                 // B*SQ rows
  float* p = wts + row * (long)SK;
  _Float16* p16 = w16 + row * (long)SK;
  const int t = threadIdx.x;
  const int lane = t & 31, wid = t >> 5;

  float x[8];
  float mx = -1e30f;
#pragma unroll
  for (int i = 0; i < 8; ++i) {
    x[i] = p[t + i * 256];
    mx = fmaxf(mx, x[i]);
  }
#pragma unroll
  for (int off = 16; off > 0; off >>= 1) mx = fmaxf(mx, __shfl_xor(mx, off, 32));
  if (lane == 0) red[wid] = mx;
  __syncthreads();
  mx = red[0];
#pragma unroll
  for (int i = 1; i < 8; ++i) mx = fmaxf(mx, red[i]);

  float s = 0.f;
#pragma unroll
  for (int i = 0; i < 8; ++i) {
    x[i] = expf(x[i] - mx);
    s += x[i];
  }
#pragma unroll
  for (int off = 16; off > 0; off >>= 1) s += __shfl_xor(s, off, 32);
  __syncthreads();
  if (lane == 0) red[wid] = s;
  __syncthreads();
  s = 0.f;
#pragma unroll
  for (int i = 0; i < 8; ++i) s += red[i];
  const float inv = 1.0f / s;
#pragma unroll
  for (int i = 0; i < 8; ++i) {
    float y = x[i] * inv;
    p[t + i * 256] = y;
    p16[t + i * 256] = (_Float16)y;
  }
}

// ---------------------------------------------------------------------------
// Kernel 3: O = W16 @ V (f16 WMMA). Wave tile 16 rows x 64 cols; all operands
// are pre-converted f16, contiguous global loads (no LDS, no conversions).
// ---------------------------------------------------------------------------
__global__ __launch_bounds__(256) void k3_pv(const _Float16* __restrict__ w16,
                                             const _Float16* __restrict__ vt,
                                             float* __restrict__ out) {
  const int NT = D / 64;                           // 8 col-blocks of 64
  const int wave = threadIdx.x >> 5;
  const int lane = threadIdx.x & 31;
  const long tile = (long)blockIdx.x * 8 + wave;   // B*(SQ/16)*NT waves
  const int b   = (int)(tile / ((SQ / 16) * NT));
  const int rem = (int)(tile % ((SQ / 16) * NT));
  const int ti = rem / NT, nb = rem % NT;

  const int m  = lane & 15;
  const int hs = lane >> 4;

  const _Float16* wrow = w16 + ((long)b * SQ + ti * 16 + m) * SK;
  const _Float16* vtb  = vt + ((long)b * D + nb * 64 + m) * SK;   // + ni*16*SK

  v8f acc[4] = {};
  for (int k0 = 0; k0 < SK; k0 += 32) {
    const _Float16* ap = wrow + k0 + hs * 8;
    v16h af = cat16(*(const v8h*)ap, *(const v8h*)(ap + 16));
#pragma unroll
    for (int ni = 0; ni < 4; ++ni) {
      const _Float16* bp = vtb + (long)ni * 16 * SK + k0 + hs * 8;
      v16h bf = cat16(*(const v8h*)bp, *(const v8h*)(bp + 16));
      acc[ni] = wmma_f16(af, bf, acc[ni]);
    }
  }

  float* ob = out + ((long)b * SQ + ti * 16) * D + nb * 64;
#pragma unroll
  for (int ni = 0; ni < 4; ++ni)
#pragma unroll
    for (int r = 0; r < 8; ++r) {
      int mm = r + 8 * hs;
      ob[(long)mm * D + ni * 16 + m] = acc[ni][r];
    }
}

// ---------------------------------------------------------------------------
extern "C" void kernel_launch(void* const* d_in, const int* in_sizes, int n_in,
                              void* d_out, int out_size, void* d_ws, size_t ws_size,
                              hipStream_t stream) {
  const float* q = (const float*)d_in[0];
  const float* k = (const float*)d_in[1];
  const float* v = (const float*)d_in[2];
  const float* t = (const float*)d_in[3];

  float* out      = (float*)d_out;
  float* attended = out;                         // [B,SQ,D]
  float* weights  = out + (long)B * SQ * D;      // [B,SQ,SK]

  // Workspace carve-up (all region sizes are large powers of two -> aligned)
  const size_t QKN = (size_t)B * SQ * D;         // 8.39M elements
  const size_t WN  = (size_t)B * SQ * SK;        // 33.6M elements
  char* ws = (char*)d_ws;
  _Float16* q_hi = (_Float16*)ws;               ws += QKN * 2;
  _Float16* q_lo = (_Float16*)ws;               ws += QKN * 2;
  _Float16* k_hi = (_Float16*)ws;               ws += QKN * 2;
  _Float16* k_lo = (_Float16*)ws;               ws += QKN * 2;
  _Float16* v_t  = (_Float16*)ws;               ws += QKN * 2;
  _Float16* w16  = (_Float16*)ws;               ws += WN * 2;
  float*    ksq  = (float*)ws;                  // B*SK floats

  const int splitBlocks = (int)(QKN / 4 / 256);  // 8192
  prep_split<<<splitBlocks, 256, 0, stream>>>(q, q_hi, q_lo);
  prep_split<<<splitBlocks, 256, 0, stream>>>(k, k_hi, k_lo);
  prep_vt<<<splitBlocks, 256, 0, stream>>>(v, v_t);
  k0_ksq<<<(B * SK) / 256, 256, 0, stream>>>(k, ksq);

  // 1) raw logits into weights region (split-f16 WMMA, 32x32 wave tiles)
  {
    long tiles = (long)B * (SQ / 32) * (SK / 32);   // 32768 waves
    k1_logits<<<(int)(tiles / 8), 256, 0, stream>>>(q_hi, q_lo, k_hi, k_lo, ksq,
                                                    t, weights);
  }
  // 2) row softmax in place + f16 weight copy
  k2_softmax<<<B * SQ, 256, 0, stream>>>(weights, w16);
  // 3) attended = w16 @ v_t (f16 WMMA, 16x64 wave tiles)
  {
    long tiles = (long)B * (SQ / 16) * (D / 64);    // 8192 waves
    k3_pv<<<(int)(tiles / 8), 256, 0, stream>>>(w16, v_t, attended);
  }
}
